// GATLayer_91122026152396
// MI455X (gfx1250) — matrix operand order repaired and checked by reading, not compiled, not run on hardware
//
#include <hip/hip_runtime.h>
#include <stdint.h>

typedef __attribute__((ext_vector_type(16))) _Float16 v16h;
typedef __attribute__((ext_vector_type(8)))  float    v8f;

#define LN_EPS 1e-5f
#define NKS  32    // neighbours per node
#define DD   128   // node feature dim
#define EE   64    // edge feature dim
#define MM   192   // D + E
#define HH   8
#define KSZ  16
#define OUTC 128
#define FFC  512

union FragH { v16h v; uint32_t u[8]; };

__device__ __forceinline__ v8f wmma16(v16h a, v16h b, v8f c) {
  // v_wmma_f32_16x16x32_f16
  return __builtin_amdgcn_wmma_f32_16x16x32_f16(false, a, false, b, (short)0, c, false, false);
}

// A fragment (16x32 f16), p -> element [0][0] of tile, lda in halfs (even).
// lanes 0-15 = rows, K0..15 in VGPR0-7; lanes 16-31 hold K16..31 (contiguous, matches load_frag_b).
__device__ __forceinline__ v16h load_frag_a(const _Float16* p, int lda) {
  int lane = threadIdx.x & 31;
  int m = lane & 15, kh = lane >> 4;
  const uint32_t* row = (const uint32_t*)(p + m * lda);
  FragH f;
#pragma unroll
  for (int j = 0; j < 4; ++j) {
    f.u[j]     = row[kh * 8 + j];
    f.u[4 + j] = row[kh * 8 + 4 + j];
  }
  return f.v;
}

// B fragment (32x16 f16): colp -> the 32 contiguous K-halfs of THIS lane's column.
// lanes 0-15 hold K0..15, lanes 16-31 hold K16..31.
__device__ __forceinline__ v16h load_frag_b(const uint32_t* colp) {
  int kh = (threadIdx.x >> 4) & 1;
  FragH f;
#pragma unroll
  for (int j = 0; j < 8; ++j) f.u[j] = colp[kh * 8 + j];
  return f.v;
}

// ---------------- LayerNorm 1 (masked) ----------------
__global__ __launch_bounds__(128)
void ln1_kernel(const float* __restrict__ node_rep, const float* __restrict__ mask,
                const float* __restrict__ s, const float* __restrict__ b,
                float* __restrict__ x32, _Float16* __restrict__ x16) {
  int n = blockIdx.x, t = threadIdx.x;
  float v = node_rep[(size_t)n * DD + t];
  float s1 = v, s2 = v * v;
#pragma unroll
  for (int off = 1; off < 32; off <<= 1) {
    s1 += __shfl_xor(s1, off, 32);
    s2 += __shfl_xor(s2, off, 32);
  }
  __shared__ float p1[4], p2[4];
  int w = t >> 5;
  if ((t & 31) == 0) { p1[w] = s1; p2[w] = s2; }
  __syncthreads();
  s1 = p1[0] + p1[1] + p1[2] + p1[3];
  s2 = p2[0] + p2[1] + p2[2] + p2[3];
  float mean = s1 * (1.0f / DD);
  float var  = s2 * (1.0f / DD) - mean * mean;
  float y = (v - mean) * rsqrtf(var + LN_EPS) * s[t] + b[t];
  y *= mask[n];
  x32[(size_t)n * DD + t] = y;
  x16[(size_t)n * DD + t] = (_Float16)y;
}

// ------------- weight convert + transpose: in f32 [R][C] -> out f16 [C][R] -------------
__global__ void transcvt_kernel(const float* __restrict__ in, _Float16* __restrict__ out,
                                int R, int C) {
  int idx = blockIdx.x * blockDim.x + threadIdx.x;
  if (idx >= R * C) return;
  int r = idx / C, c = idx % C;
  out[(size_t)c * R + r] = (_Float16)in[idx];
}

// ---------------- generic WMMA GEMM: A[M][K] f16  x  Bt[Ncols][K] f16 ----------------
// block = 128 threads (4 waves), tile 64 rows x 64 cols, wave w -> rows [16w,16w+16)
// EPI 0: f32 out = acc + bias
// EPI 1: f16 out = relu(acc + bias)
// EPI 2: f32 out = addend + mask[row] * (acc + bias)
template <int EPI>
__global__ __launch_bounds__(128)
void gemm_kernel(const _Float16* __restrict__ A, int Mrows, int Kdim,
                 const _Float16* __restrict__ Bt, const float* __restrict__ bias,
                 float* __restrict__ of32, _Float16* __restrict__ of16,
                 const float* __restrict__ addend, const float* __restrict__ mask, int ldo) {
  __shared__ _Float16 As[64][32];
  __shared__ _Float16 Bs[64][32];  // column-major: [col][k]
  int t = threadIdx.x;
  int w = t >> 5, lane = t & 31;
  int row0 = blockIdx.x * 64;
  int col0 = blockIdx.y * 64;
  v8f zero = {};
  v8f acc[4];
#pragma unroll
  for (int i = 0; i < 4; ++i) acc[i] = zero;

  int lr = t >> 1, seg = t & 1;
  for (int k0 = 0; k0 < Kdim; k0 += 32) {
    __syncthreads();
    uint4 va0 = {0, 0, 0, 0}, va1 = {0, 0, 0, 0};
    int grow = row0 + lr;
    if (grow < Mrows) {
      const uint4* pa = (const uint4*)(A + (size_t)grow * Kdim + k0 + seg * 16);
      va0 = pa[0]; va1 = pa[1];
    }
    *(uint4*)&As[lr][seg * 16]     = va0;
    *(uint4*)&As[lr][seg * 16 + 8] = va1;
    const uint4* pb = (const uint4*)(Bt + (size_t)(col0 + lr) * Kdim + k0 + seg * 16);
    *(uint4*)&Bs[lr][seg * 16]     = pb[0];
    *(uint4*)&Bs[lr][seg * 16 + 8] = pb[1];
    __syncthreads();

    v16h af = load_frag_a(&As[w * 16][0], 32);
#pragma unroll
    for (int ct = 0; ct < 4; ++ct) {
      v16h bf = load_frag_b((const uint32_t*)&Bs[ct * 16 + (lane & 15)][0]);
      acc[ct] = wmma16(af, bf, acc[ct]);
    }
  }

  int n = lane & 15, kh = lane >> 4;
#pragma unroll
  for (int ct = 0; ct < 4; ++ct) {
    int gcol = col0 + ct * 16 + n;
    float bv = bias[gcol];
#pragma unroll
    for (int r = 0; r < 8; ++r) {
      int grow = row0 + w * 16 + kh * 8 + r;
      if (grow >= Mrows) continue;
      float val = acc[ct][r] + bv;
      if (EPI == 0) {
        of32[(size_t)grow * ldo + gcol] = val;
      } else if (EPI == 1) {
        of16[(size_t)grow * ldo + gcol] = (_Float16)fmaxf(val, 0.0f);
      } else {
        of32[(size_t)grow * ldo + gcol] =
            addend[(size_t)grow * ldo + gcol] + mask[grow] * val;
      }
    }
  }
}

// ---------------- fused per-node attention ----------------
// 1 block per node, 128 threads (4 waves). kk/vv via WMMA, then softmax + o@Wo + LN2.
__global__ __launch_bounds__(128)
void attn_kernel(const _Float16* __restrict__ x16, const float* __restrict__ qf,
                 const float* __restrict__ edge, const int* __restrict__ senders,
                 const _Float16* __restrict__ Wkt, const _Float16* __restrict__ Wvt,
                 const float* __restrict__ bk, const float* __restrict__ bv,
                 const _Float16* __restrict__ Wot, const float* __restrict__ bo,
                 const float* __restrict__ maskp,
                 const float* __restrict__ ln2s, const float* __restrict__ ln2b,
                 _Float16* __restrict__ t16) {
  __shared__ _Float16 Ash[NKS][MM];   // 32 x 192 neighbour matrix
  __shared__ float kksh[NKS][OUTC];   // keys   32 x 128
  __shared__ float vvsh[NKS][OUTC];   // values 32 x 128
  __shared__ float attw[HH][NKS];
  __shared__ float osh[OUTC];
  __shared__ float p1[4], p2[4];

  int node = blockIdx.x;
  int t = threadIdx.x;
  int w = t >> 5, lane = t & 31;

  // stage: Ash[k] = [ x16[senders[k]] (128 halfs) | f16(edge[k]) (64 halfs) ]
  {
    int k = t >> 2, part = t & 3;
    int s = senders[node * NKS + k];
    const uint4* src = (const uint4*)(x16 + (size_t)s * DD + part * 32);
    uint4* dst = (uint4*)&Ash[k][part * 32];
#pragma unroll
    for (int j = 0; j < 4; ++j) dst[j] = src[j];
    const float* es = edge + ((size_t)node * NKS + k) * EE + part * 16;
    _Float16* ed = &Ash[k][DD + part * 16];
#pragma unroll
    for (int j = 0; j < 16; ++j) ed[j] = (_Float16)es[j];
  }
  __syncthreads();

  // kk/vv projection: wave w produces output columns [32w, 32w+32)
  {
    v8f zero = {};
    v8f accK[2][2], accV[2][2];
#pragma unroll
    for (int a = 0; a < 2; ++a)
#pragma unroll
      for (int b = 0; b < 2; ++b) { accK[a][b] = zero; accV[a][b] = zero; }
    int n = lane & 15;
#pragma unroll
    for (int kt = 0; kt < MM / 32; ++kt) {
      v16h a0 = load_frag_a(&Ash[0][kt * 32], MM);
      v16h a1 = load_frag_a(&Ash[16][kt * 32], MM);
#pragma unroll
      for (int ctl = 0; ctl < 2; ++ctl) {
        int col = (2 * w + ctl) * 16 + n;
        v16h bkf = load_frag_b((const uint32_t*)(Wkt + (size_t)col * MM + kt * 32));
        v16h bvf = load_frag_b((const uint32_t*)(Wvt + (size_t)col * MM + kt * 32));
        accK[0][ctl] = wmma16(a0, bkf, accK[0][ctl]);
        accK[1][ctl] = wmma16(a1, bkf, accK[1][ctl]);
        accV[0][ctl] = wmma16(a0, bvf, accV[0][ctl]);
        accV[1][ctl] = wmma16(a1, bvf, accV[1][ctl]);
      }
    }
    int kh = lane >> 4;
#pragma unroll
    for (int ctl = 0; ctl < 2; ++ctl) {
      int col = (2 * w + ctl) * 16 + n;
      float bkv = bk[col], bvv = bv[col];
#pragma unroll
      for (int rt = 0; rt < 2; ++rt)
#pragma unroll
        for (int r = 0; r < 8; ++r) {
          int row = rt * 16 + kh * 8 + r;
          kksh[row][col] = accK[rt][ctl][r] + bkv;
          vvsh[row][col] = accV[rt][ctl][r] + bvv;
        }
    }
  }
  __syncthreads();

  int h = t >> 4, i = t & 15;   // head h lives in one 16-lane half-wave
  // logits + softmax over K=32 (each lane owns neighbours i and i+16)
  {
    float l0 = 0.f, l1 = 0.f;
    const float* qrow = qf + (size_t)node * DD + h * KSZ;
#pragma unroll
    for (int d = 0; d < KSZ; ++d) {
      float qd = qrow[d];
      l0 += qd * kksh[i][h * KSZ + d];
      l1 += qd * kksh[i + 16][h * KSZ + d];
    }
    l0 *= 0.25f; l1 *= 0.25f;   // 1/sqrt(KS)
    float mx = fmaxf(l0, l1);
#pragma unroll
    for (int off = 1; off < 16; off <<= 1) mx = fmaxf(mx, __shfl_xor(mx, off, 32));
    float e0 = __expf(l0 - mx), e1 = __expf(l1 - mx);
    float s = e0 + e1;
#pragma unroll
    for (int off = 1; off < 16; off <<= 1) s += __shfl_xor(s, off, 32);
    float inv = 1.0f / s;
    attw[h][i] = e0 * inv;
    attw[h][i + 16] = e1 * inv;
  }
  __syncthreads();
  // o[h][i] = sum_k attw[h][k] * vv[k][h*16+i]
  {
    float a = 0.f;
#pragma unroll
    for (int k = 0; k < NKS; ++k) a += attw[h][k] * vvsh[k][h * KSZ + i];
    osh[t] = a;
  }
  __syncthreads();
  // res = o @ Wo + bo ; mask ; LN2 -> f16
  float maskv = maskp[node];
  float r = 0.f;
  const _Float16* wrow = Wot + (size_t)t * OUTC;
  for (int j = 0; j < OUTC; ++j) r += osh[j] * (float)wrow[j];
  r = (r + bo[t]) * maskv;
  float s1 = r, s2 = r * r;
#pragma unroll
  for (int off = 1; off < 32; off <<= 1) {
    s1 += __shfl_xor(s1, off, 32);
    s2 += __shfl_xor(s2, off, 32);
  }
  if (lane == 0) { p1[w] = s1; p2[w] = s2; }
  __syncthreads();
  s1 = p1[0] + p1[1] + p1[2] + p1[3];
  s2 = p2[0] + p2[1] + p2[2] + p2[3];
  float mean = s1 * (1.0f / OUTC);
  float var  = s2 * (1.0f / OUTC) - mean * mean;
  float y = (r - mean) * rsqrtf(var + LN_EPS) * ln2s[t] + ln2b[t];
  t16[(size_t)node * OUTC + t] = (_Float16)y;
}

extern "C" void kernel_launch(void* const* d_in, const int* in_sizes, int n_in,
                              void* d_out, int out_size, void* d_ws, size_t ws_size,
                              hipStream_t stream) {
  (void)n_in; (void)out_size; (void)ws_size;
  const int N = in_sizes[0] / DD;  // 20000
  const float* node_rep = (const float*)d_in[0];
  const float* mask     = (const float*)d_in[1];
  const float* edge     = (const float*)d_in[2];
  const float* Wq = (const float*)d_in[3];
  const float* bq = (const float*)d_in[4];
  const float* Wk = (const float*)d_in[5];
  const float* bk = (const float*)d_in[6];
  const float* Wv = (const float*)d_in[7];
  const float* bv = (const float*)d_in[8];
  const float* Wo = (const float*)d_in[9];
  const float* bo = (const float*)d_in[10];
  const float* ln1s = (const float*)d_in[11];
  const float* ln1b = (const float*)d_in[12];
  const float* ln2s = (const float*)d_in[13];
  const float* ln2b = (const float*)d_in[14];
  const float* W1 = (const float*)d_in[15];
  const float* b1 = (const float*)d_in[16];
  const float* W2 = (const float*)d_in[17];
  const float* b2 = (const float*)d_in[18];
  const int* senders = (const int*)d_in[19];

  char* ws = (char*)d_ws;
  size_t off = 0;
  auto alloc = [&](size_t bytes) -> void* {
    void* p = ws + off;
    off = (off + bytes + 255) & ~(size_t)255;
    return p;
  };
  float*    x32 = (float*)alloc((size_t)N * DD * 4);
  _Float16* x16 = (_Float16*)alloc((size_t)N * DD * 2);
  float*    qf  = (float*)alloc((size_t)N * DD * 4);
  _Float16* t16 = (_Float16*)alloc((size_t)N * OUTC * 2);
  _Float16* h16 = (_Float16*)alloc((size_t)N * FFC * 2);
  _Float16* Wqt = (_Float16*)alloc((size_t)DD * DD * 2);
  _Float16* Wkt = (_Float16*)alloc((size_t)MM * DD * 2);
  _Float16* Wvt = (_Float16*)alloc((size_t)MM * DD * 2);
  _Float16* Wot = (_Float16*)alloc((size_t)DD * OUTC * 2);
  _Float16* W1t = (_Float16*)alloc((size_t)OUTC * FFC * 2);
  _Float16* W2t = (_Float16*)alloc((size_t)FFC * OUTC * 2);

  ln1_kernel<<<N, 128, 0, stream>>>(node_rep, mask, ln1s, ln1b, x32, x16);
  transcvt_kernel<<<(DD * DD + 255) / 256, 256, 0, stream>>>(Wq, Wqt, DD, DD);
  transcvt_kernel<<<(MM * DD + 255) / 256, 256, 0, stream>>>(Wk, Wkt, MM, DD);
  transcvt_kernel<<<(MM * DD + 255) / 256, 256, 0, stream>>>(Wv, Wvt, MM, DD);
  transcvt_kernel<<<(DD * OUTC + 255) / 256, 256, 0, stream>>>(Wo, Wot, DD, OUTC);
  transcvt_kernel<<<(OUTC * FFC + 255) / 256, 256, 0, stream>>>(W1, W1t, OUTC, FFC);
  transcvt_kernel<<<(FFC * OUTC + 255) / 256, 256, 0, stream>>>(W2, W2t, FFC, OUTC);

  dim3 gq((N + 63) / 64, DD / 64);
  gemm_kernel<0><<<gq, 128, 0, stream>>>(x16, N, DD, Wqt, bq, qf, nullptr, nullptr, nullptr, DD);

  attn_kernel<<<N, 128, 0, stream>>>(x16, qf, edge, senders, Wkt, Wvt, bk, bv,
                                     Wot, bo, mask, ln2s, ln2b, t16);

  dim3 g1((N + 63) / 64, FFC / 64);
  gemm_kernel<1><<<g1, 128, 0, stream>>>(t16, N, OUTC, W1t, b1, nullptr, h16, nullptr, nullptr, FFC);

  dim3 g2((N + 63) / 64, OUTC / 64);
  gemm_kernel<2><<<g2, 128, 0, stream>>>(h16, N, FFC, W2t, b2, (float*)d_out, nullptr, x32, mask, OUTC);
}
